// Router_33724083208470
// MI455X (gfx1250) — matrix operand order repaired and checked by reading, not compiled, not run on hardware
//
#include <hip/hip_runtime.h>
#include <hip/hip_fp16.h>

// ---------------------------------------------------------------------------
// Router MLP for MI455X (gfx1250, wave32, WMMA f16 16x16x32)
//   B=4096, N_SLOTS=64, D_SLOT=D_WORK=1024, H1=512, H2=256, M=16, TAU=1
// ---------------------------------------------------------------------------

typedef __attribute__((ext_vector_type(16))) _Float16 v16h;
typedef __attribute__((ext_vector_type(8)))  float    v8f;
typedef __attribute__((ext_vector_type(4)))  _Float16 v4h;
typedef __attribute__((ext_vector_type(4)))  float    v4f;

#define B_SZ    4096
#define DSLOT   1024
#define DWORK   1024
#define DIN     2048
#define H1_SZ   512
#define H2_SZ   256
#define M_SZ    16

__device__ __forceinline__ float gelu_exact(float x) {
    return 0.5f * x * (1.0f + erff(x * 0.7071067811865476f));
}

// A fragment (16x32 f16, MxK): lane (m&15) holds row m; K offsets
// {d..d+7, d+16..d+23}, d = (lane>=16) ? 8 : 0.  rowbase = &A[row][0].
__device__ __forceinline__ v16h load_a_frag(const _Float16* __restrict__ rowbase,
                                            int k0, int lane) {
    const _Float16* p = rowbase + k0 + ((lane & 16) ? 8 : 0);
    v16h f;
    ((v4f*)&f)[0] = *(const v4f*)(p);
    ((v4f*)&f)[1] = *(const v4f*)(p + 16);
    return f;
}

// B fragment (32x16 f16, KxN) where B = W^T and W is row-major [N][K]:
// lane (n&15) holds column n of B = row n of W; K offsets
// {d..d+7, d+16..d+23}, d = (lane>=16) ? 8 : 0 (same as A).
// rowbase = &W[n][0].
__device__ __forceinline__ v16h load_b_frag(const _Float16* __restrict__ rowbase,
                                            int k0, int lane) {
    const _Float16* p = rowbase + k0 + ((lane & 16) ? 8 : 0);
    v16h f;
    ((v4f*)&f)[0] = *(const v4f*)(p);
    ((v4f*)&f)[1] = *(const v4f*)(p + 16);
    return f;
}

// ---------------------------------------------------------------------------
// Kernel 0: convert W1 [512x2048] and W2 [256x512] fp32 -> f16
// ---------------------------------------------------------------------------
__global__ __launch_bounds__(256) void k_convert_weights(
        const float* __restrict__ W1, const float* __restrict__ W2,
        _Float16* __restrict__ w1h, _Float16* __restrict__ w2h) {
    const int n1 = H1_SZ * DIN;        // 1048576
    const int n2 = H2_SZ * H1_SZ;      // 131072
    int i = (blockIdx.x * blockDim.x + threadIdx.x) * 4;
    if (i < n1) {
        v4f v = *(const v4f*)(W1 + i);
        v4h h = { (_Float16)v.x, (_Float16)v.y, (_Float16)v.z, (_Float16)v.w };
        *(v4h*)(w1h + i) = h;
    } else if (i - n1 < n2) {
        int j = i - n1;
        v4f v = *(const v4f*)(W2 + j);
        v4h h = { (_Float16)v.x, (_Float16)v.y, (_Float16)v.z, (_Float16)v.w };
        *(v4h*)(w2h + j) = h;
    }
}

// ---------------------------------------------------------------------------
// Kernel 1: mean-pool slots over N_SLOTS + concat working_mem -> x f16 [B,2048]
// One block per batch row; block sweeps 64 contiguous 4KB rows.
// Slots (1.07 GB) are streamed once: non-temporal loads keep them out of the
// L2 retention set (weights/x/h1 stay resident instead).
// ---------------------------------------------------------------------------
__global__ __launch_bounds__(256) void k_pool_concat(
        const float* __restrict__ slots, const float* __restrict__ wm,
        _Float16* __restrict__ x) {
    const int b  = blockIdx.x;
    const int d4 = threadIdx.x;                    // 0..255 -> cols 4*d4
    const float* sp = slots + ((size_t)b * 64) * DSLOT + d4 * 4;
    v4f acc = {0.f, 0.f, 0.f, 0.f};
    #pragma unroll 8
    for (int n = 0; n < 64; ++n) {
        v4f v = __builtin_nontemporal_load((const v4f*)(sp + (size_t)n * DSLOT));
        acc += v;
    }
    const float s = 1.0f / 64.0f;
    v4h hp = { (_Float16)(acc.x * s), (_Float16)(acc.y * s),
               (_Float16)(acc.z * s), (_Float16)(acc.w * s) };
    _Float16* xrow = x + (size_t)b * DIN;
    *(v4h*)(xrow + d4 * 4) = hp;

    v4f w = *(const v4f*)(wm + (size_t)b * DWORK + d4 * 4);
    v4h hw = { (_Float16)w.x, (_Float16)w.y, (_Float16)w.z, (_Float16)w.w };
    *(v4h*)(xrow + DSLOT + d4 * 4) = hw;
}

// ---------------------------------------------------------------------------
// Kernel 2: h1 = gelu(LN(x @ W1^T + b1))   [4096x512] f16
// Block = 16-row M-tile x full N=512; 8 waves x 4 WMMA tiles; 64 K-steps.
// Ping-pong double buffering (2 K-steps per loop iter) -> no register
// rotation copies, loads fly under the WMMAs with partial loadcnt waits.
// ---------------------------------------------------------------------------
__global__ __launch_bounds__(256) void k_gemm1_ln_gelu(
        const _Float16* __restrict__ x, const _Float16* __restrict__ w1h,
        const float* __restrict__ b1, const float* __restrict__ gamma,
        const float* __restrict__ beta, _Float16* __restrict__ h1) {
    __shared__ float ylds[16][H1_SZ + 4];
    const int m0   = blockIdx.x * 16;
    const int wave = threadIdx.x >> 5;
    const int lane = threadIdx.x & 31;
    const int n0   = wave * 64;

    const _Float16* arow  = x   + (size_t)(m0 + (lane & 15)) * DIN;
    const _Float16* brow0 = w1h + (size_t)(n0 + (lane & 15)) * DIN;

    v8f acc[4] = {v8f{}, v8f{}, v8f{}, v8f{}};

    auto LOADF = [&](int k0, v16h& a, v16h (&b)[4]) {
        a = load_a_frag(arow, k0, lane);
        #pragma unroll
        for (int t = 0; t < 4; ++t)
            b[t] = load_b_frag(brow0 + t * 16 * DIN, k0, lane);
    };
    auto MMA = [&](const v16h& a, const v16h (&b)[4]) {
        #pragma unroll
        for (int t = 0; t < 4; ++t)
            acc[t] = __builtin_amdgcn_wmma_f32_16x16x32_f16(
                         false, a, false, b[t], (short)0, acc[t], false, false);
    };

    v16h aA, aB;
    v16h bA[4], bB[4];
    LOADF(0, aA, bA);
    // 64 K-steps total: 1 (prologue) + 31*2 (loop) + 1 (epilogue)
    #pragma unroll 1
    for (int k0 = 32; k0 < DIN - 32; k0 += 64) {
        LOADF(k0, aB, bB);
        MMA(aA, bA);
        LOADF(k0 + 32, aA, bA);
        MMA(aB, bB);
    }
    LOADF(DIN - 32, aB, bB);
    MMA(aA, bA);
    MMA(aB, bB);

    // D layout: VGPR r -> rows r (lanes 0-15) / r+8 (lanes 16-31), N = lane&15
    const int rbase = (lane & 16) ? 8 : 0;
    #pragma unroll
    for (int t = 0; t < 4; ++t) {
        int col = n0 + t * 16 + (lane & 15);
        #pragma unroll
        for (int r = 0; r < 8; ++r)
            ylds[rbase + r][col] = acc[t][r];
    }
    __syncthreads();

    // bias + LayerNorm(512) + exact GELU; 16 threads per row
    const int row = threadIdx.x >> 4;
    const int sc  = threadIdx.x & 15;
    float xv[32];
    float sum = 0.f, ssq = 0.f;
    #pragma unroll
    for (int j = 0; j < 32; ++j) {
        int c = sc + j * 16;
        float v = ylds[row][c] + b1[c];
        xv[j] = v; sum += v; ssq += v * v;
    }
    #pragma unroll
    for (int off = 1; off < 16; off <<= 1) {
        sum += __shfl_xor(sum, off, 16);
        ssq += __shfl_xor(ssq, off, 16);
    }
    const float mean = sum * (1.0f / 512.0f);
    const float var  = ssq * (1.0f / 512.0f) - mean * mean;
    const float rstd = rsqrtf(var + 1e-5f);
    _Float16* orow = h1 + (size_t)(m0 + row) * H1_SZ;
    #pragma unroll
    for (int j = 0; j < 32; ++j) {
        int c = sc + j * 16;
        float v = (xv[j] - mean) * rstd * gamma[c] + beta[c];
        orow[c] = (_Float16)gelu_exact(v);
    }
}

// ---------------------------------------------------------------------------
// Kernel 3: h2 = gelu(h1 @ W2^T + b2); logits = h2 @ W3^T + b3 + gumbel;
//           gates = softmax(logits)  -> d_out [4096x16] f32
// Block = 16-row M-tile x full N=256; 8 waves x 2 WMMA tiles; 16 K-steps.
// ---------------------------------------------------------------------------
__global__ __launch_bounds__(256) void k_gemm2_gemm3_softmax(
        const _Float16* __restrict__ h1, const _Float16* __restrict__ w2h,
        const float* __restrict__ b2, const float* __restrict__ W3,
        const float* __restrict__ b3, const float* __restrict__ gumbel,
        float* __restrict__ gates) {
    // pad = 4 floats keeps every row 16-byte aligned (260*4 = 1040 = 65*16)
    __shared__ float h2lds[16][H2_SZ + 4];
    __shared__ float w3lds[M_SZ][H2_SZ + 4];
    const int m0   = blockIdx.x * 16;
    const int lane = threadIdx.x & 31;
    const int wave = threadIdx.x >> 5;
    const int n0   = wave * 32;

    // stage W3 (16x256 fp32) into padded LDS
    #pragma unroll
    for (int j = 0; j < 16; ++j) {
        int idx = threadIdx.x + j * 256;
        w3lds[idx >> 8][idx & 255] = W3[idx];
    }

    const _Float16* arow = h1  + (size_t)(m0 + (lane & 15)) * H1_SZ;
    const _Float16* brow = w2h + (size_t)(n0 + (lane & 15)) * H1_SZ;

    v8f acc[2] = {v8f{}, v8f{}};

    auto LOADF = [&](int k0, v16h& a, v16h (&b)[2]) {
        a = load_a_frag(arow, k0, lane);
        #pragma unroll
        for (int t = 0; t < 2; ++t)
            b[t] = load_b_frag(brow + t * 16 * H1_SZ, k0, lane);
    };
    auto MMA = [&](const v16h& a, const v16h (&b)[2]) {
        #pragma unroll
        for (int t = 0; t < 2; ++t)
            acc[t] = __builtin_amdgcn_wmma_f32_16x16x32_f16(
                         false, a, false, b[t], (short)0, acc[t], false, false);
    };

    v16h aA, aB;
    v16h bA[2], bB[2];
    LOADF(0, aA, bA);
    // 16 K-steps total: 1 + 7*2 + 1
    #pragma unroll 1
    for (int k0 = 32; k0 < H1_SZ - 32; k0 += 64) {
        LOADF(k0, aB, bB);
        MMA(aA, bA);
        LOADF(k0 + 32, aA, bA);
        MMA(aB, bB);
    }
    LOADF(H1_SZ - 32, aB, bB);
    MMA(aA, bA);
    MMA(aB, bB);

    const int rbase = (lane & 16) ? 8 : 0;
    #pragma unroll
    for (int t = 0; t < 2; ++t) {
        int col = n0 + t * 16 + (lane & 15);
        float bias = b2[col];
        #pragma unroll
        for (int r = 0; r < 8; ++r)
            h2lds[rbase + r][col] = gelu_exact(acc[t][r] + bias);
    }
    __syncthreads();

    // GEMM3 (K=256) + gumbel-softmax; thread = (row, logit), float4 LDS reads
    const int row  = threadIdx.x >> 4;
    const int o    = threadIdx.x & 15;
    const int grow = m0 + row;
    float logit = b3[o] + gumbel[grow * M_SZ + o];
    #pragma unroll 4
    for (int k = 0; k < H2_SZ; k += 4) {
        v4f hv = *(const v4f*)(&h2lds[row][k]);
        v4f wv = *(const v4f*)(&w3lds[o][k]);
        logit += hv.x * wv.x + hv.y * wv.y + hv.z * wv.z + hv.w * wv.w;
    }

    float mx = logit;
    #pragma unroll
    for (int off = 1; off < 16; off <<= 1)
        mx = fmaxf(mx, __shfl_xor(mx, off, 16));
    float e = expf(logit - mx);
    float s = e;
    #pragma unroll
    for (int off = 1; off < 16; off <<= 1)
        s += __shfl_xor(s, off, 16);
    gates[grow * M_SZ + o] = e / s;
}

// ---------------------------------------------------------------------------
// Inputs (setup_inputs order): slots, working_mem, gumbel_noise,
//   W1, b1, gamma, beta, W2, b2, W3, b3   (all fp32)
// ---------------------------------------------------------------------------
extern "C" void kernel_launch(void* const* d_in, const int* in_sizes, int n_in,
                              void* d_out, int out_size, void* d_ws, size_t ws_size,
                              hipStream_t stream) {
    (void)in_sizes; (void)n_in; (void)out_size; (void)ws_size;
    const float* slots  = (const float*)d_in[0];
    const float* wm     = (const float*)d_in[1];
    const float* gumbel = (const float*)d_in[2];
    const float* W1     = (const float*)d_in[3];
    const float* b1     = (const float*)d_in[4];
    const float* gamma  = (const float*)d_in[5];
    const float* beta   = (const float*)d_in[6];
    const float* W2     = (const float*)d_in[7];
    const float* b2     = (const float*)d_in[8];
    const float* W3     = (const float*)d_in[9];
    const float* b3     = (const float*)d_in[10];
    float* gates = (float*)d_out;

    // workspace layout (f16): x [4096x2048], W1h [512x2048], W2h [256x512],
    //                         h1 [4096x512]   -> ~22.3 MB total
    char* ws = (char*)d_ws;
    _Float16* xh  = (_Float16*)(ws);
    _Float16* w1h = (_Float16*)(ws + (size_t)B_SZ * DIN * 2);
    _Float16* w2h = (_Float16*)(ws + (size_t)B_SZ * DIN * 2 + (size_t)H1_SZ * DIN * 2);
    _Float16* h1  = (_Float16*)(ws + (size_t)B_SZ * DIN * 2 + (size_t)H1_SZ * DIN * 2
                                   + (size_t)H2_SZ * H1_SZ * 2);

    const int nconv = (H1_SZ * DIN + H2_SZ * H1_SZ) / 4;      // 294912
    k_convert_weights<<<(nconv + 255) / 256, 256, 0, stream>>>(W1, W2, w1h, w2h);
    k_pool_concat<<<B_SZ, 256, 0, stream>>>(slots, wm, xh);
    k_gemm1_ln_gelu<<<B_SZ / 16, 256, 0, stream>>>(xh, w1h, b1, gamma, beta, h1);
    k_gemm2_gemm3_softmax<<<B_SZ / 16, 256, 0, stream>>>(h1, w2h, b2, W3, b3, gumbel, gates);
}
